// MultiHeadedAttention_2688649528075
// MI455X (gfx1250) — matrix operand order repaired and checked by reading, not compiled, not run on hardware
//
#include <hip/hip_runtime.h>
#include <hip/hip_fp16.h>

// ---------------------------------------------------------------------------
// Windowed MHA (B=4, C=128, H=W=256, NH=4, DK=32, 8x8 windows) + 3x3 reflect
// conv + LeakyReLU.  All matmul stages via v_wmma_f32_16x16x32_f16 (wave32).
// ---------------------------------------------------------------------------

#define B_   4
#define C_   128
#define H_   256
#define W_   256
#define NH_  4
#define NW_  1024          // tokens (windows) per (b, head) = 32*32
#define DD_  2048          // per-token feature dim = DK * 8 * 8
#define QELE ((size_t)B_ * NH_ * NW_ * DD_)   // elements per q/k/v tensor
#define SCALE_ 0.02209708691207961f           // 1/sqrt(2048)

typedef __attribute__((ext_vector_type(16))) _Float16 v16h;
typedef __attribute__((ext_vector_type(8)))  _Float16 v8h;
typedef __attribute__((ext_vector_type(8)))  float    v8f;
typedef __attribute__((ext_vector_type(4)))  float    v4f;

// ---- forced-global (addrspace 1) accessors: avoid flat_load/flat_store -----
typedef __attribute__((address_space(1))) const v16h*     gc16;
typedef __attribute__((address_space(1))) const v8h*      gc8;
typedef __attribute__((address_space(1))) v8h*            g8;
typedef __attribute__((address_space(1))) const _Float16* gch;
typedef __attribute__((address_space(1))) _Float16*       gh;
typedef __attribute__((address_space(1))) const float*    gcf;
typedef __attribute__((address_space(1))) v4f*            g4f;

__device__ __forceinline__ v16h gload16(const _Float16* p) {
  return *(gc16)(unsigned long long)p;
}
__device__ __forceinline__ v8h gload8(const _Float16* p) {
  return *(gc8)(unsigned long long)p;
}
__device__ __forceinline__ void gstore8(_Float16* p, v8h v) {
  *(g8)(unsigned long long)p = v;
}
__device__ __forceinline__ _Float16 gloadh(const _Float16* p) {
  return *(gch)(unsigned long long)p;
}
__device__ __forceinline__ void gstoreh(_Float16* p, _Float16 v) {
  *(gh)(unsigned long long)p = v;
}
__device__ __forceinline__ float gloadf(const float* p) {
  return *(gcf)(unsigned long long)p;
}
__device__ __forceinline__ void gstore_nt4(float* p, v4f v) {
  __builtin_nontemporal_store(v, (g4f)(unsigned long long)p);
}

__device__ __forceinline__ v8f wmma_f16(const v16h& a, const v16h& b, const v8f& c) {
  // D = A(16x32 f16) * B(32x16 f16) + C(16x16 f32)
  return __builtin_amdgcn_wmma_f32_16x16x32_f16(false, a, false, b, (short)0, c,
                                                false, false);
}

// A-matrix 16x32 f16 fragment (k = 2*hb + s, contiguous): caller passes pointer to this
// lane's M-row at the K-slice base; hb = (lane&16)>>1 (0 or 8).
__device__ __forceinline__ v16h frag_a(const _Float16* __restrict__ row, int hb) {
  v16h a;
#pragma unroll
  for (int p = 0; p < 8; ++p) {
    int kb = (hb << 1) + (p << 1);
    a[2 * p]     = row[kb];
    a[2 * p + 1] = row[kb + 1];
  }
  return a;
}

__device__ __forceinline__ v8f vzero8() {
  v8f z = {0.f, 0.f, 0.f, 0.f, 0.f, 0.f, 0.f, 0.f};
  return z;
}

// ---------------------------------------------------------------------------
// K0: one-time repacks (all f32 -> f16, WMMA-friendly layouts):
//   wqkv16[mat][cout][cin]            (3*128*128)
//   bqkv  [mat][cout]  (f32)          (3*128)
//   wof16 [cout][tap*128+cin]         (128*1152)
// ---------------------------------------------------------------------------
__global__ void repack_kernel(const float* __restrict__ wq,
                              const float* __restrict__ bq,
                              const float* __restrict__ wk,
                              const float* __restrict__ bk,
                              const float* __restrict__ wv,
                              const float* __restrict__ bv,
                              const float* __restrict__ wo,
                              _Float16* __restrict__ wqkv16,
                              float* __restrict__ bqkv,
                              _Float16* __restrict__ wof16) {
  int i = blockIdx.x * 256 + threadIdx.x;
  if (i < 3 * 128 * 128) {                    // qkv weights
    int mat = i >> 14, r = i & 16383;
    const float* src = (mat == 0) ? wq : (mat == 1) ? wk : wv;
    wqkv16[i] = (_Float16)gloadf(src + r);
    if (r < 128) {                            // biases
      const float* bs = (mat == 0) ? bq : (mat == 1) ? bk : bv;
      bqkv[mat * 128 + r] = gloadf(bs + r);
    }
    return;
  }
  int j = i - 3 * 128 * 128;
  if (j < 128 * 1152) {                       // conv weights, tap-major K
    int cout = j / 1152;
    int r    = j - cout * 1152;
    int t    = r >> 7;                        // tap 0..8
    int cin  = r & 127;
    wof16[j] = (_Float16)gloadf(wo + ((size_t)cout * 128 + cin) * 9 + t);
  }
}

// ---------------------------------------------------------------------------
// K1: fused Q/K/V 1x1 conv projection, written directly in windowed layout
//     qkv[mat][b][nh][token][dk*64 + py*8 + px]  (f16).
// One workgroup per (b, window): GEMM 64 px x 128 cout, K=128.
// ---------------------------------------------------------------------------
__global__ __launch_bounds__(256) void qkv_kernel(
    const float* __restrict__ x,
    const _Float16* __restrict__ wqkv,   // [3][128][128] f16
    const float* __restrict__ bqkv,      // [3][128] f32
    _Float16* __restrict__ qkv) {        // q | k | v, each QELE elements
  __shared__ _Float16 sA[64 * 132];      // [pixel][cin] f16, padded
  int b   = blockIdx.x >> 10;
  int win = blockIdx.x & 1023;           // token = wy*32+wx
  int wy = win >> 5, wx = win & 31;
  int tid = threadIdx.x, lane = tid & 31, wave = tid >> 5;

  const float* xb = x + (size_t)b * C_ * H_ * W_;
#pragma unroll
  for (int i = 0; i < 32; ++i) {
    int idx = i * 256 + tid;             // 0..8191
    int cin = idx >> 6, pix = idx & 63;
    int h = wy * 8 + (pix >> 3), w = wx * 8 + (pix & 7);
    sA[pix * 132 + cin] = (_Float16)gloadf(xb + ((size_t)cin * H_ + h) * W_ + w);
  }
  __syncthreads();

  int n  = lane & 15;
  int hb = (lane & 16) >> 1;             // 0 / 8
  int kb16 = (lane >> 4) << 4;           // 0 / 16
  int cout = wave * 16 + n;              // 8 waves cover 128 couts
  int nh = cout >> 5, dk = cout & 31;
  size_t obase = (((size_t)(b * NH_ + nh)) * NW_ + win) * DD_ + dk * 64;

  for (int mat = 0; mat < 3; ++mat) {
    const _Float16* wm = wqkv + mat * (128 * 128) + (size_t)cout * C_;
    v8f acc[4];
#pragma unroll
    for (int mt = 0; mt < 4; ++mt) acc[mt] = vzero8();

#pragma unroll
    for (int ks = 0; ks < 4; ++ks) {
      v16h bf = gload16(wm + ks * 32 + kb16);   // contiguous 32B
#pragma unroll
      for (int mt = 0; mt < 4; ++mt) {
        v16h af = frag_a(sA + (mt * 16 + n) * 132 + ks * 32, hb);
        acc[mt] = wmma_f16(af, bf, acc[mt]);
      }
    }
    float bias = gloadf(bqkv + mat * 128 + cout);
    _Float16* op = qkv + (size_t)mat * QELE + obase;
#pragma unroll
    for (int mt = 0; mt < 4; ++mt) {
      int mb = mt * 16 + hb;             // pixel base, contiguous 8 in d
      v8h hv;
#pragma unroll
      for (int r = 0; r < 8; ++r) hv[r] = (_Float16)(acc[mt][r] + bias);
      gstore8(op + mb, hv);              // 16B aligned global store
    }
  }
}

// ---------------------------------------------------------------------------
// K2: scores = QK^T / sqrt(d), row softmax -> P (f16).
// One workgroup per (b, head, 16-query block). Score row kept in LDS.
// ---------------------------------------------------------------------------
__global__ __launch_bounds__(256) void attn_scores_kernel(
    const _Float16* __restrict__ qw, const _Float16* __restrict__ kwin,
    _Float16* __restrict__ Pm) {
  extern __shared__ char smemRaw[];
  _Float16* sQ = (_Float16*)smemRaw;                // 16 x 2056 f16
  float*    sS = (float*)(smemRaw + 16 * 2056 * 2); // 16 x 1028 f32

  int bh = blockIdx.x >> 6;                         // (b*NH+nh)
  int qb = blockIdx.x & 63;
  int tid = threadIdx.x, lane = tid & 31, wave = tid >> 5;
  const _Float16* Q = qw + ((size_t)bh * NW_ + qb * 16) * DD_;
  const _Float16* Km = kwin + (size_t)bh * NW_ * DD_;

  // stage the 16x2048 Q tile (bank-conflict pad 2048->2056)
#pragma unroll
  for (int i = 0; i < 16; ++i) {
    int l = (i * 256 + tid) * 8;        // half index 0..32767
    int row = l >> 11, col = l & 2047;
    *(v8h*)(sQ + row * 2056 + col) = gload8(Q + (size_t)row * 2048 + col);
  }
  __syncthreads();

  int n = lane & 15, hb = (lane & 16) >> 1, kb16 = (lane >> 4) << 4;
  v8f acc[8];
#pragma unroll
  for (int nt = 0; nt < 8; ++nt) acc[nt] = vzero8();

  for (int ks = 0; ks < 64; ++ks) {                 // K = 2048
    v16h af = frag_a(sQ + n * 2056 + ks * 32, hb);
#pragma unroll
    for (int nt = 0; nt < 8; ++nt) {                // wave covers 128 keys
      int key = (wave * 8 + nt) * 16 + n;
      v16h bf = gload16(Km + (size_t)key * 2048 + ks * 32 + kb16);
      acc[nt] = wmma_f16(af, bf, acc[nt]);
    }
  }
#pragma unroll
  for (int nt = 0; nt < 8; ++nt) {
    float* dst = sS + (wave * 8 + nt) * 16 + n;
#pragma unroll
    for (int r = 0; r < 8; ++r) dst[(hb + r) * 1028] = acc[nt][r] * SCALE_;
  }
  __syncthreads();

  // softmax: each wave owns 2 of the 16 rows (row length 1024)
  _Float16* Pp = Pm + ((size_t)bh * NW_ + qb * 16) * 1024;
#pragma unroll
  for (int rr = 0; rr < 2; ++rr) {
    int row = wave * 2 + rr;
    float vals[32];
    float mx = -3.4e38f;
#pragma unroll
    for (int i = 0; i < 32; ++i) {
      vals[i] = sS[row * 1028 + lane + i * 32];
      mx = fmaxf(mx, vals[i]);
    }
#pragma unroll
    for (int off = 16; off; off >>= 1) mx = fmaxf(mx, __shfl_xor(mx, off, 32));
    float sum = 0.f;
#pragma unroll
    for (int i = 0; i < 32; ++i) {
      vals[i] = __expf(vals[i] - mx);
      sum += vals[i];
    }
#pragma unroll
    for (int off = 16; off; off >>= 1) sum += __shfl_xor(sum, off, 32);
    float inv = 1.f / sum;
#pragma unroll
    for (int i = 0; i < 32; ++i)
      gstoreh(Pp + (size_t)row * 1024 + lane + i * 32,
              (_Float16)(vals[i] * inv));
  }
}

// ---------------------------------------------------------------------------
// K3: Y = P @ V, stored as planar f16 y[b][c][h][w] for the conv stage.
// Workgroup tile: 64 queries x 128 d-cols, K = 1024 keys.
// ---------------------------------------------------------------------------
__global__ __launch_bounds__(256) void attn_pv_kernel(
    const _Float16* __restrict__ Pm, const _Float16* __restrict__ vw,
    _Float16* __restrict__ yh) {
  __shared__ _Float16 sP[64 * 40];     // [query][key]  (32 + pad)
  __shared__ _Float16 sV[128 * 40];    // [d][key] transposed

  int bh  = blockIdx.x >> 8;
  int rem = blockIdx.x & 255;
  int mb = rem >> 4, nb = rem & 15;
  int qbase = mb * 64, nbase = nb * 128;
  int b = bh >> 2, nh = bh & 3;
  int tid = threadIdx.x, lane = tid & 31, wave = tid >> 5;
  const _Float16* P = Pm + ((size_t)bh * NW_ + qbase) * 1024;
  const _Float16* V = vw + (size_t)bh * NW_ * DD_ + nbase;

  int n = lane & 15, hb = (lane & 16) >> 1, kb16 = (lane >> 4) << 4;
  v8f acc[4];
#pragma unroll
  for (int mt = 0; mt < 4; ++mt) acc[mt] = vzero8();

  for (int ks = 0; ks < 32; ++ks) {
    __syncthreads();
    {  // P tile 64x32, coalesced 16B chunks
      int row = tid >> 2, col = (tid & 3) * 8;
      *(v8h*)(sP + row * 40 + col) =
          gload8(P + (size_t)row * 1024 + ks * 32 + col);
    }
    {  // V tile 32 keys x 128 d, transposed into sV[d][key]
      int key = tid >> 3, dcol = (tid & 7) * 16;
      v16h vv = gload16(V + (size_t)(ks * 32 + key) * 2048 + dcol);
#pragma unroll
      for (int j = 0; j < 16; ++j) sV[(dcol + j) * 40 + key] = vv[j];
    }
    __syncthreads();
    v16h bf = *(const v16h*)(sV + (wave * 16 + n) * 40 + kb16);
#pragma unroll
    for (int mt = 0; mt < 4; ++mt) {
      v16h af = frag_a(sP + (mt * 16 + n) * 40, hb);
      acc[mt] = wmma_f16(af, bf, acc[mt]);
    }
  }

  // scatter back to planar layout: d -> (dk, py, px), token -> (oy, ox)
  int d = nbase + wave * 16 + n;
  int dk = d >> 6, p = d & 63, py = p >> 3, px = p & 7;
  int c = nh * 32 + dk;
#pragma unroll
  for (int mt = 0; mt < 4; ++mt) {
#pragma unroll
    for (int r = 0; r < 8; ++r) {
      int q = qbase + mt * 16 + hb + r;
      int oy = q >> 5, ox = q & 31;
      int h = oy * 8 + py, w = ox * 8 + px;
      gstoreh(yh + (((size_t)b * C_ + c) * H_ + h) * W_ + w,
              (_Float16)acc[mt][r]);
    }
  }
}

// ---------------------------------------------------------------------------
// K4: 3x3 conv (reflect pad) + bias + LeakyReLU(0.2) via implicit GEMM.
// Workgroup = (b, 8x8 pixel block); halo tile 10x10x128 in LDS; K = 1152
// ordered tap-major so each 32-wide K-step stays inside one tap.
// ---------------------------------------------------------------------------
__global__ __launch_bounds__(256) void conv3_kernel(
    const _Float16* __restrict__ yh, const _Float16* __restrict__ wof16,
    const float* __restrict__ bo, float* __restrict__ out) {
  __shared__ _Float16 sX[100 * 132];   // [iy*10+ix][cin]

  int b  = blockIdx.x >> 10;
  int sb = blockIdx.x & 1023;
  int by = sb >> 5, bx = sb & 31;
  int y0 = by * 8, x0 = bx * 8;
  int tid = threadIdx.x, lane = tid & 31, wave = tid >> 5;

  for (int i = tid; i < 12800; i += 256) {
    int c = i / 100;
    int s = i - c * 100;
    int iy = s / 10, ix = s - iy * 10;
    int h = y0 - 1 + iy; h = h < 0 ? -h : (h > 255 ? 510 - h : h);
    int w = x0 - 1 + ix; w = w < 0 ? -w : (w > 255 ? 510 - w : w);
    sX[s * 132 + c] = gloadh(yh + (((size_t)b * C_ + c) * H_ + h) * W_ + w);
  }
  __syncthreads();

  int n = lane & 15, hb = (lane & 16) >> 1, kb16 = (lane >> 4) << 4;
  int cout = wave * 16 + n;
  v8f acc[4];
#pragma unroll
  for (int mt = 0; mt < 4; ++mt) acc[mt] = vzero8();

  for (int ks = 0; ks < 36; ++ks) {    // K = 9 taps * 128 cin
    int t = ks >> 2;
    int cinb = (ks & 3) << 5;
    int dy = t / 3, dx = t - dy * 3;   // 0..2 (halo already offsets by -1)
    v16h bf = gload16(wof16 + (size_t)cout * 1152 + ks * 32 + kb16);
#pragma unroll
    for (int mt = 0; mt < 4; ++mt) {
      int p = mt * 16 + n;             // output pixel index in 8x8 block
      int s = ((p >> 3) + dy) * 10 + (p & 7) + dx;
      v16h af = frag_a(sX + s * 132 + cinb, hb);
      acc[mt] = wmma_f16(af, bf, acc[mt]);
    }
  }

  float bias = gloadf(bo + cout);
#pragma unroll
  for (int mt = 0; mt < 4; ++mt) {
    int pb = mt * 16 + hb;             // multiple of 8 -> one pixel row
    int py = pb >> 3;
    float v[8];
#pragma unroll
    for (int r = 0; r < 8; ++r) {
      float t = acc[mt][r] + bias;
      v[r] = t > 0.f ? t : 0.2f * t;
    }
    float* dst = out + (((size_t)b * C_ + cout) * H_ + y0 + py) * W_ + x0;
    v4f lo = {v[0], v[1], v[2], v[3]};
    v4f hi = {v[4], v[5], v[6], v[7]};
    gstore_nt4(dst, lo);               // no reuse of `out` -> NT hint
    gstore_nt4(dst + 4, hi);
  }
}

// ---------------------------------------------------------------------------
extern "C" void kernel_launch(void* const* d_in, const int* in_sizes, int n_in,
                              void* d_out, int out_size, void* d_ws,
                              size_t ws_size, hipStream_t stream) {
  const float* x  = (const float*)d_in[0];
  // d_in[1] = mask m: the reference's masked_fill is a no-op -> unused
  const float* wq = (const float*)d_in[2];
  const float* bq = (const float*)d_in[3];
  const float* wk = (const float*)d_in[4];
  const float* bk = (const float*)d_in[5];
  const float* wv = (const float*)d_in[6];
  const float* bv = (const float*)d_in[7];
  const float* wo = (const float*)d_in[8];
  const float* bo = (const float*)d_in[9];
  float* out = (float*)d_out;

  // workspace partition (~302 MB total, all 256B aligned)
  char* ws = (char*)d_ws;
  size_t o = 0;
  _Float16* qkv = (_Float16*)(ws + o); o += 3 * QELE * 2;                    // 192 MB (q|k|v)
  _Float16* Pm  = (_Float16*)(ws + o); o += (size_t)B_ * NH_ * NW_ * NW_ * 2; // 32 MB
  _Float16* yh  = (_Float16*)(ws + o); o += (size_t)B_ * C_ * H_ * W_ * 2;    // 64 MB
  _Float16* wof16  = (_Float16*)(ws + o); o += (size_t)128 * 1152 * 2;
  _Float16* wqkv16 = (_Float16*)(ws + o); o += (size_t)3 * 128 * 128 * 2;
  float*    bqkv   = (float*)(ws + o);

  _Float16* qw = qkv;
  _Float16* kw = qkv + QELE;
  _Float16* vw = qkv + 2 * QELE;

  int repack_n = 3 * 128 * 128 + 128 * 1152;
  repack_kernel<<<(repack_n + 255) / 256, 256, 0, stream>>>(
      wq, bq, wk, bk, wv, bv, wo, wqkv16, bqkv, wof16);
  qkv_kernel<<<B_ * 1024, 256, 0, stream>>>(x, wqkv16, bqkv, qkv);
  attn_scores_kernel<<<B_ * NH_ * 64, 256,
                       16 * 2056 * 2 + 16 * 1028 * 4, stream>>>(qw, kw, Pm);
  attn_pv_kernel<<<B_ * NH_ * 256, 256, 0, stream>>>(Pm, vw, yh);
  conv3_kernel<<<B_ * 1024, 256, 0, stream>>>(yh, wof16, bo, out);
}
